// BitNetDecoder_47785806135532
// MI455X (gfx1250) — 8-bit matrix operand order repaired and checked by reading, compiled, not run on hardware
//
#include <hip/hip_runtime.h>
#include <cstdint>
#include <cstddef>

#define EPSF 1e-5f

typedef __attribute__((ext_vector_type(8))) int v8i;
typedef int v4i_ __attribute__((vector_size(16)));

#if __has_builtin(__builtin_amdgcn_global_load_async_to_lds_b128)
#define HAVE_ASYNC_LDS 1
#else
#define HAVE_ASYNC_LDS 0
#endif

#define AS1 __attribute__((address_space(1)))
#define AS3 __attribute__((address_space(3)))

static constexpr int BATCH = 4096;
static constexpr int D_IN  = 1024;
static constexpr int HDIM  = 4096;
static constexpr int NOUT  = 2000;
static constexpr int BINS  = 1000;

__device__ __forceinline__ void async_wait0() {
#if HAVE_ASYNC_LDS
#if __has_builtin(__builtin_amdgcn_s_wait_asynccnt)
    __builtin_amdgcn_s_wait_asynccnt(0);
#else
    asm volatile("s_wait_asynccnt 0x0" ::: "memory");
#endif
#endif
}

__global__ void wabs_partial_kernel(const float* __restrict__ W,
                                    float* __restrict__ partial, long n) {
    __shared__ float sred[256];
    float s = 0.f;
    for (long i = blockIdx.x * 256L + threadIdx.x; i < n; i += 256L * gridDim.x)
        s += fabsf(W[i]);
    sred[threadIdx.x] = s;
    __syncthreads();
    for (int st = 128; st > 0; st >>= 1) {
        if ((int)threadIdx.x < st) sred[threadIdx.x] += sred[threadIdx.x + st];
        __syncthreads();
    }
    if (threadIdx.x == 0) partial[blockIdx.x] = sred[0];
}

__global__ void wabs_final_kernel(const float* __restrict__ partial,
                                  float* __restrict__ wdq, long n, int nparts) {
    __shared__ float sred[256];
    float s = 0.f;
    for (int i = threadIdx.x; i < nparts; i += 256) s += partial[i];
    sred[threadIdx.x] = s;
    __syncthreads();
    for (int st = 128; st > 0; st >>= 1) {
        if ((int)threadIdx.x < st) sred[threadIdx.x] += sred[threadIdx.x + st];
        __syncthreads();
    }
    if (threadIdx.x == 0) {
        float mean = sred[0] / (float)n;
        wdq[0] = fmaxf(mean, EPSF);
    }
}

__global__ void wquant_kernel(const float* __restrict__ W,
                              const float* __restrict__ wdq,
                              signed char* __restrict__ Wq, long n4) {
    const float scale = 1.0f / wdq[0];
    long i = blockIdx.x * 256L + threadIdx.x;
    if (i >= n4) return;
    float4 w = ((const float4*)W)[i];
    int qx = (int)rintf(w.x * scale); qx = qx < -1 ? -1 : (qx > 1 ? 1 : qx);
    int qy = (int)rintf(w.y * scale); qy = qy < -1 ? -1 : (qy > 1 ? 1 : qy);
    int qz = (int)rintf(w.z * scale); qz = qz < -1 ? -1 : (qz > 1 ? 1 : qz);
    int qw = (int)rintf(w.w * scale); qw = qw < -1 ? -1 : (qw > 1 ? 1 : qw);
    char4 q; q.x = (signed char)qx; q.y = (signed char)qy;
    q.z = (signed char)qz; q.w = (signed char)qw;
    ((char4*)Wq)[i] = q;
}

__global__ void act_quant_kernel(const float* __restrict__ X,
                                 signed char* __restrict__ Xq,
                                 float* __restrict__ xs) {
    constexpr int D = D_IN, PER = D / 256;
    __shared__ float sred[256];
    const int row = blockIdx.x, t = threadIdx.x;
    const float* xr = X + (size_t)row * D;
    float vals[PER];
    float amax = 0.f;
#pragma unroll
    for (int j = 0; j < PER; ++j) {
        float v = xr[t + 256 * j];
        vals[j] = v;
        amax = fmaxf(amax, fabsf(v));
    }
    sred[t] = amax; __syncthreads();
    for (int st = 128; st > 0; st >>= 1) {
        if (t < st) sred[t] = fmaxf(sred[t], sred[t + st]);
        __syncthreads();
    }
    amax = fmaxf(sred[0], EPSF);
    const float scale = 127.f / amax;
#pragma unroll
    for (int j = 0; j < PER; ++j) {
        int q = (int)rintf(vals[j] * scale);
        q = q < -128 ? -128 : (q > 127 ? 127 : q);
        Xq[(size_t)row * D + t + 256 * j] = (signed char)q;
    }
    if (t == 0) xs[row] = amax * (1.0f / 127.f);
}

__global__ void ln_silu_quant_kernel(const float* __restrict__ C,
                                     const float* __restrict__ g,
                                     const float* __restrict__ b,
                                     signed char* __restrict__ Xq,
                                     float* __restrict__ xs) {
    constexpr int D = HDIM, PER = D / 256;
    __shared__ float sred[256];
    const int row = blockIdx.x, t = threadIdx.x;
    const float* cr = C + (size_t)row * D;
    float vals[PER];
    float s = 0.f;
#pragma unroll
    for (int j = 0; j < PER; ++j) { vals[j] = cr[t + 256 * j]; s += vals[j]; }
    sred[t] = s; __syncthreads();
    for (int st = 128; st > 0; st >>= 1) {
        if (t < st) sred[t] += sred[t + st];
        __syncthreads();
    }
    const float mean = sred[0] / (float)D;
    __syncthreads();
    s = 0.f;
#pragma unroll
    for (int j = 0; j < PER; ++j) { float d = vals[j] - mean; s += d * d; }
    sred[t] = s; __syncthreads();
    for (int st = 128; st > 0; st >>= 1) {
        if (t < st) sred[t] += sred[t + st];
        __syncthreads();
    }
    const float rstd = rsqrtf(sred[0] / (float)D + EPSF);
    __syncthreads();
    float amax = 0.f;
#pragma unroll
    for (int j = 0; j < PER; ++j) {
        const int idx = t + 256 * j;
        float y = (vals[j] - mean) * rstd * g[idx] + b[idx];
        float si = y / (1.f + expf(-y));
        vals[j] = si;
        amax = fmaxf(amax, fabsf(si));
    }
    sred[t] = amax; __syncthreads();
    for (int st = 128; st > 0; st >>= 1) {
        if (t < st) sred[t] = fmaxf(sred[t], sred[t + st]);
        __syncthreads();
    }
    amax = fmaxf(sred[0], EPSF);
    const float scale = 127.f / amax;
#pragma unroll
    for (int j = 0; j < PER; ++j) {
        int q = (int)rintf(vals[j] * scale);
        q = q < -128 ? -128 : (q > 127 ? 127 : q);
        Xq[(size_t)row * D + t + 256 * j] = (signed char)q;
    }
    if (t == 0) xs[row] = amax * (1.0f / 127.f);
}

__global__ __launch_bounds__(256)
void gemm_iu8_wmma_kernel(const signed char* __restrict__ Aq,
                          const float* __restrict__ ascl,
                          const signed char* __restrict__ Wq,
                          const float* __restrict__ wdqp,
                          float* __restrict__ C,
                          int M, int N, int K) {
    __shared__ __align__(16) signed char As[2][128 * 64];
    __shared__ __align__(16) signed char Bs[2][128 * 64];

    const int t     = threadIdx.x;
    const int lane  = t & 31;
    const int wave  = t >> 5;
    const int waveM = wave & 1;
    const int waveN = wave >> 1;
    const int m0    = blockIdx.y * 128;
    const int n0    = blockIdx.x * 128;
    const int mrow  = lane & 15;
    const int half  = lane >> 4;
    const float wdq = wdqp[0];

    v8i acc[4][2] = {};

    const int ldRow = t >> 1;
    const int ldCol = (t & 1) * 32;
    const int aRow  = m0 + ldRow;
    int bRow        = n0 + ldRow;
    if (bRow >= N) bRow = N - 1;

    const int nIter = K >> 6;

    auto stage = [&](int bufIdx, int kk) {
        const signed char* agp = Aq + (size_t)aRow * K + kk + ldCol;
        const signed char* bgp = Wq + (size_t)bRow * K + kk + ldCol;
        signed char* asp = &As[bufIdx][ldRow * 64 + ldCol];
        signed char* bsp = &Bs[bufIdx][ldRow * 64 + ldCol];
#if HAVE_ASYNC_LDS
        __builtin_amdgcn_global_load_async_to_lds_b128(
            (AS1 v4i_*)agp, (AS3 v4i_*)asp, 0, 0);
        __builtin_amdgcn_global_load_async_to_lds_b128(
            (AS1 v4i_*)agp, (AS3 v4i_*)asp, 16, 0);
        __builtin_amdgcn_global_load_async_to_lds_b128(
            (AS1 v4i_*)bgp, (AS3 v4i_*)bsp, 0, 0);
        __builtin_amdgcn_global_load_async_to_lds_b128(
            (AS1 v4i_*)bgp, (AS3 v4i_*)bsp, 16, 0);
#else
        int4 a0 = ((const int4*)agp)[0];
        int4 a1 = ((const int4*)agp)[1];
        int4 b0 = ((const int4*)bgp)[0];
        int4 b1 = ((const int4*)bgp)[1];
        ((int4*)asp)[0] = a0; ((int4*)asp)[1] = a1;
        ((int4*)bsp)[0] = b0; ((int4*)bsp)[1] = b1;
        __builtin_prefetch(agp + 64, 0, 3);
        __builtin_prefetch(bgp + 64, 0, 3);
#endif
    };

    stage(0, 0);

    for (int it = 0; it < nIter; ++it) {
        const int buf = it & 1;
        async_wait0();
        __syncthreads();
        if (it + 1 < nIter) stage(buf ^ 1, (it + 1) << 6);

        v8i bf[2];
#pragma unroll
        for (int nt = 0; nt < 2; ++nt) {
            const int c = waveN * 32 + nt * 16 + mrow;
            int4 d0 = *(const int4*)(&Bs[buf][c * 64 + half * 16]);
            int4 d1 = *(const int4*)(&Bs[buf][c * 64 + 32 + half * 16]);
            bf[nt][0] = d0.x; bf[nt][1] = d0.y; bf[nt][2] = d0.z; bf[nt][3] = d0.w;
            bf[nt][4] = d1.x; bf[nt][5] = d1.y; bf[nt][6] = d1.z; bf[nt][7] = d1.w;
        }
#pragma unroll
        for (int mt = 0; mt < 4; ++mt) {
            const int r = waveM * 64 + mt * 16 + mrow;
            v8i af;
#pragma unroll
            for (int p = 0; p < 2; ++p) {
                int4 d = *(const int4*)(&As[buf][r * 64 + p * 32 + half * 16]);
                af[4 * p] = d.x; af[4 * p + 1] = d.y; af[4 * p + 2] = d.z; af[4 * p + 3] = d.w;
            }
#pragma unroll
            for (int nt = 0; nt < 2; ++nt) {
                acc[mt][nt] = __builtin_amdgcn_wmma_i32_16x16x64_iu8(
                    true, af, true, bf[nt], acc[mt][nt], false, false);
            }
        }
        __syncthreads();
    }

#pragma unroll
    for (int mt = 0; mt < 4; ++mt) {
#pragma unroll
        for (int v = 0; v < 8; ++v) {
            const int grow = m0 + waveM * 64 + mt * 16 + v + 8 * half;
            const float s = ascl[grow] * wdq;
#pragma unroll
            for (int nt = 0; nt < 2; ++nt) {
                const int gcol = n0 + waveN * 32 + nt * 16 + mrow;
                if (gcol < N)
                    C[(size_t)grow * N + gcol] = (float)acc[mt][nt][v] * s;
            }
        }
    }
}

__global__ void head_kernel(const float* __restrict__ C, float* __restrict__ out) {
    long i = blockIdx.x * 256L + threadIdx.x;
    if (i >= (long)BATCH * NOUT) return;
    const int row = (int)(i / NOUT);
    const int col = (int)(i % NOUT);
    const float sig = 1.f / (1.f + expf(-C[i]));
    if (col < BINS)
        out[(size_t)row * BINS + col] = sig * (float)(BINS - 1) + 1.f;
    else
        out[(size_t)BATCH * BINS + (size_t)row * BINS + (col - BINS)] = sig * 100.f;
}

extern "C" void kernel_launch(void* const* d_in, const int* in_sizes, int n_in,
                              void* d_out, int out_size, void* d_ws, size_t ws_size,
                              hipStream_t stream) {
    (void)in_sizes; (void)n_in; (void)out_size; (void)ws_size;

    const float* x  = (const float*)d_in[0];
    const float* W0 = (const float*)d_in[1];
    const float* W1 = (const float*)d_in[2];
    const float* W2 = (const float*)d_in[3];
    const float* W3 = (const float*)d_in[4];
    const float* g0 = (const float*)d_in[5];
    const float* b0 = (const float*)d_in[6];
    const float* g1 = (const float*)d_in[7];
    const float* b1 = (const float*)d_in[8];
    const float* g2 = (const float*)d_in[9];
    const float* b2 = (const float*)d_in[10];

    char* ws = (char*)d_ws;
    size_t off = 0;
    auto alloc = [&](size_t bytes) -> char* {
        char* p = ws + off;
        off += (bytes + 255) & ~(size_t)255;
        return p;
    };

    signed char* qW0 = (signed char*)alloc((size_t)HDIM * D_IN);
    signed char* qW1 = (signed char*)alloc((size_t)HDIM * HDIM);
    signed char* qW2 = (signed char*)alloc((size_t)HDIM * HDIM);
    signed char* qW3 = (signed char*)alloc((size_t)NOUT * HDIM);
    signed char* Xq  = (signed char*)alloc((size_t)BATCH * HDIM);
    float* xs        = (float*)alloc((size_t)BATCH * sizeof(float));
    float* wdq       = (float*)alloc(4 * sizeof(float));
    float* partial   = (float*)alloc(1024 * sizeof(float));
    float* Cbuf      = (float*)alloc((size_t)BATCH * HDIM * sizeof(float));

    const float* Ws[4]  = {W0, W1, W2, W3};
    signed char* qWs[4] = {qW0, qW1, qW2, qW3};
    const long   wn[4]  = {(long)HDIM * D_IN, (long)HDIM * HDIM,
                           (long)HDIM * HDIM, (long)NOUT * HDIM};

    for (int i = 0; i < 4; ++i) {
        wabs_partial_kernel<<<1024, 256, 0, stream>>>(Ws[i], partial, wn[i]);
        wabs_final_kernel<<<1, 256, 0, stream>>>(partial, wdq + i, wn[i], 1024);
        long n4 = wn[i] / 4;
        wquant_kernel<<<(int)((n4 + 255) / 256), 256, 0, stream>>>(Ws[i], wdq + i,
                                                                   qWs[i], n4);
    }

    act_quant_kernel<<<BATCH, 256, 0, stream>>>(x, Xq, xs);
    gemm_iu8_wmma_kernel<<<dim3(HDIM / 128, BATCH / 128), 256, 0, stream>>>(
        Xq, xs, qW0, wdq + 0, Cbuf, BATCH, HDIM, D_IN);
    ln_silu_quant_kernel<<<BATCH, 256, 0, stream>>>(Cbuf, g0, b0, Xq, xs);

    gemm_iu8_wmma_kernel<<<dim3(HDIM / 128, BATCH / 128), 256, 0, stream>>>(
        Xq, xs, qW1, wdq + 1, Cbuf, BATCH, HDIM, HDIM);
    ln_silu_quant_kernel<<<BATCH, 256, 0, stream>>>(Cbuf, g1, b1, Xq, xs);

    gemm_iu8_wmma_kernel<<<dim3(HDIM / 128, BATCH / 128), 256, 0, stream>>>(
        Xq, xs, qW2, wdq + 2, Cbuf, BATCH, HDIM, HDIM);
    ln_silu_quant_kernel<<<BATCH, 256, 0, stream>>>(Cbuf, g2, b2, Xq, xs);

    gemm_iu8_wmma_kernel<<<dim3((NOUT + 127) / 128, BATCH / 128), 256, 0, stream>>>(
        Xq, xs, qW3, wdq + 3, Cbuf, BATCH, NOUT, HDIM);

    head_kernel<<<(int)(((long)BATCH * NOUT + 255) / 256), 256, 0, stream>>>(
        Cbuf, (float*)d_out);
}
